// MultiHeadAttention_55611236549114
// MI455X (gfx1250) — compile-verified, matrix operand order checked by reading, not run on hardware
//
#include <hip/hip_runtime.h>
#include <hip/hip_bf16.h>

// ---------------------------------------------------------------------------
// Problem constants (match reference)
// ---------------------------------------------------------------------------
#define B_    2
#define S_    2048
#define HID_  2048
#define NH_   16
#define NKV_  4
#define HD_   128
#define GRP_  (NH_ / NKV_)       // 4
#define M_    (B_ * S_)          // 4096 rows for all GEMMs
#define NQKV_ 5120               // [Wq|Wk|Wv] fused: 4096 (q+gate) + 512 (k) + 512 (v)

typedef unsigned short u16;
typedef __attribute__((ext_vector_type(8)))  u16          u16x8;
typedef __attribute__((ext_vector_type(16))) u16          u16x16;
typedef __attribute__((ext_vector_type(16))) __bf16       bf16x16;
typedef __attribute__((ext_vector_type(8)))  float        f32x8;
typedef __attribute__((ext_vector_type(4)))  unsigned int u32x4;
typedef __attribute__((ext_vector_type(8)))  int          i32x8;
typedef __attribute__((ext_vector_type(4)))  int          i32x4;

#if __has_builtin(__builtin_amdgcn_tensor_load_to_lds) && \
    __has_builtin(__builtin_amdgcn_s_wait_tensorcnt)
#define USE_TDM 1
#else
#define USE_TDM 0
#endif

__device__ __forceinline__ u16 f2bf(float f) {
  unsigned u = __float_as_uint(f);
  u += 0x7FFFu + ((u >> 16) & 1u);          // round-to-nearest-even
  return (u16)(u >> 16);
}

__device__ __forceinline__ bf16x16 as_bf(u16x16 v) {
  union { u16x16 u; bf16x16 b; } c; c.u = v; return c.b;
}

// Fragment gather for 16-bit A/B operands of v_wmma_f32_16x16x32_bf16.
// ISA lane layout (16-bit A 16x32): lanes 0-15 hold K={0..7,16..23},
// lanes 16-31 hold K={8..15,24..31} -> per lane: two contiguous 8-element
// (16-byte) chunks at kgrp*8 and 16+kgrp*8 => two b128 loads.
__device__ __forceinline__ u16x16 ld_frag(const u16* row, int kgrp) {
  u16x8 lo = *(const u16x8*)(row + kgrp * 8);
  u16x8 hi = *(const u16x8*)(row + 16 + kgrp * 8);
  return __builtin_shufflevector(lo, hi, 0,1,2,3,4,5,6,7,8,9,10,11,12,13,14,15);
}

__device__ __forceinline__ f32x8 wmma_bf(u16x16 a, u16x16 b, f32x8 c) {
  return __builtin_amdgcn_wmma_f32_16x16x32_bf16(false, as_bf(a), false, as_bf(b),
                                                 (short)0, c, false, false);
}

// ---------------------------------------------------------------------------
// 1) f32 -> bf16 strided convert (also used to fuse Wq|Wk|Wv side by side)
// ---------------------------------------------------------------------------
__global__ void cvt_f32_bf16(const float* __restrict__ src, u16* __restrict__ dst,
                             int rows, int cols, int sld, int dld, int doff) {
  long idx = (long)blockIdx.x * blockDim.x + threadIdx.x;
  long total = (long)rows * cols;
  if (idx >= total) return;
  int r = (int)(idx / cols), c = (int)(idx % cols);
  dst[(long)r * dld + doff + c] = f2bf(src[(long)r * sld + c]);
}

// ---------------------------------------------------------------------------
// 2) bf16 WMMA GEMM: C[M,N] = A[M,K] @ B[K,N], f32 accumulate.
//    Workgroup tile 64x128, 8 waves (2 along M x 4 along N), wave tile 32x32.
// ---------------------------------------------------------------------------
__global__ __launch_bounds__(256)
void gemm_bf16(const u16* __restrict__ A, const u16* __restrict__ Bm,
               float* __restrict__ C, int M, int N, int K,
               int lda, int ldb, int ldc) {
  __shared__ u16 ldsA[64 * 32];    // [row][k]  (row-major)
  __shared__ u16 ldsB[128 * 32];   // [n][k]    (N-major => B-fragment friendly)

  const int tid  = threadIdx.x;
  const int wave = tid >> 5;
  const int lane = tid & 31;
  const int l16  = lane & 15;
  const int kgrp = lane >> 4;

  const int mtile = blockIdx.y * 64;
  const int ntile = blockIdx.x * 128;
  const int waveM = (wave & 1) * 32;
  const int waveN = (wave >> 1) * 32;

  f32x8 acc[2][2];
  #pragma unroll
  for (int a = 0; a < 2; a++)
    #pragma unroll
    for (int b = 0; b < 2; b++)
      #pragma unroll
      for (int i = 0; i < 8; i++) acc[a][b][i] = 0.0f;

  for (int k0 = 0; k0 < K; k0 += 32) {
    // Stage A tile 64x32: one b128 per thread
    {
      int row = tid >> 2, kp = (tid & 3) * 8;
      u16x8 v = *(const u16x8*)(A + (long)(mtile + row) * lda + k0 + kp);
      *(u16x8*)(&ldsA[row * 32 + kp]) = v;
      if (k0 + 32 < K)
        __builtin_prefetch((const void*)(A + (long)(mtile + row) * lda + k0 + 32 + kp), 0, 0);
    }
    // Stage B tile 32x128 transposed into [n][k]
    #pragma unroll
    for (int i = 0; i < 2; i++) {
      int k  = (tid >> 4) + i * 16;
      int np = (tid & 15) * 8;
      u16x8 v = *(const u16x8*)(Bm + (long)(k0 + k) * ldb + ntile + np);
      #pragma unroll
      for (int j = 0; j < 8; j++) ldsB[(np + j) * 32 + k] = v[j];
    }
    __syncthreads();

    u16x16 af[2], bf[2];
    #pragma unroll
    for (int mm = 0; mm < 2; mm++)
      af[mm] = ld_frag(&ldsA[(waveM + mm * 16 + l16) * 32], kgrp);
    #pragma unroll
    for (int nn = 0; nn < 2; nn++)
      bf[nn] = ld_frag(&ldsB[(waveN + nn * 16 + l16) * 32], kgrp);

    #pragma unroll
    for (int mm = 0; mm < 2; mm++)
      #pragma unroll
      for (int nn = 0; nn < 2; nn++)
        acc[mm][nn] = wmma_bf(af[mm], bf[nn], acc[mm][nn]);
    __syncthreads();
  }

  // Epilogue: C/D layout -> m = r + 8*(lane>>4), n = lane&15
  #pragma unroll
  for (int mm = 0; mm < 2; mm++)
    #pragma unroll
    for (int nn = 0; nn < 2; nn++)
      #pragma unroll
      for (int r = 0; r < 8; r++) {
        int m = mtile + waveM + mm * 16 + r + 8 * kgrp;
        int n = ntile + waveN + nn * 16 + l16;
        C[(long)m * ldc + n] = acc[mm][nn][r];
      }
}

// ---------------------------------------------------------------------------
// 3) RMS-norm + partial RoPE (first 32 dims) + pack to bf16 head layouts.
//    grid = (B*S, NH+2*NKV), block = 128 (one head-vector per block)
// ---------------------------------------------------------------------------
__global__ __launch_bounds__(128)
void normrope_pack(const float* __restrict__ qkv,
                   const float* __restrict__ qnw, const float* __restrict__ knw,
                   u16* __restrict__ qb, u16* __restrict__ kb, u16* __restrict__ vb) {
  __shared__ float red[128];
  const int bs  = blockIdx.x;
  const int b   = bs / S_;
  const int s   = bs % S_;
  const int y   = blockIdx.y;
  const int tid = threadIdx.x;

  int col; u16* dst; const float* w = nullptr; bool do_rope = true;
  if (y < NH_) {                       // q head
    col = y * HD_ + tid;  w = qnw;
    dst = qb + ((long)(b * NH_ + y) * S_ + s) * HD_ + tid;
  } else if (y < NH_ + NKV_) {         // k head
    int h = y - NH_;
    col = 4096 + h * HD_ + tid;  w = knw;
    dst = kb + ((long)(b * NKV_ + h) * S_ + s) * HD_ + tid;
  } else {                             // v head: convert only
    int h = y - NH_ - NKV_;
    col = 4608 + h * HD_ + tid;  do_rope = false;
    dst = vb + ((long)(b * NKV_ + h) * S_ + s) * HD_ + tid;
  }

  float x = qkv[(long)bs * NQKV_ + col];
  if (!do_rope) { *dst = f2bf(x); return; }

  // RMS norm over the 128-wide head
  red[tid] = x * x;
  __syncthreads();
  for (int o = 64; o > 0; o >>= 1) {
    if (tid < o) red[tid] += red[tid + o];
    __syncthreads();
  }
  float var = red[0] * (1.0f / HD_);
  __syncthreads();
  float xn = x * rsqrtf(var + 1e-6f) * w[tid];

  // RoPE on dims [0,32): rotate-half within the 32-dim slice
  red[tid] = xn;
  __syncthreads();
  float out = xn;
  if (tid < 32) {
    int i = tid & 15;
    float inv = __powf(10000.0f, -(float)(2 * i) / 32.0f);
    float ang = (float)s * inv;
    float other = (tid < 16) ? -red[tid + 16] : red[tid - 16];
    out = xn * __cosf(ang) + other * __sinf(ang);
  }
  *dst = f2bf(out);
}

// ---------------------------------------------------------------------------
// 4) Flash attention (causal, GQA). grid = (S/64, NH, B), block = 128 (4 waves).
//    Each wave owns 16 q-rows; K tile DMA'd into LDS by the Tensor Data Mover,
//    V^T tile staged cooperatively (transpose), 32 keys per iteration.
// ---------------------------------------------------------------------------
__global__ __launch_bounds__(128)
void flash_attn(const u16* __restrict__ qb, const u16* __restrict__ kb,
                const u16* __restrict__ vb, float* __restrict__ attnf) {
  __shared__ u16 ldsK [32 * 128];      // [key][hd]
  __shared__ u16 ldsVt[128 * 32];      // [hd][key]
  __shared__ u16 ldsP [4][16 * 32];    // per-wave P tile [m][key]

  const int tid  = threadIdx.x;
  const int wave = tid >> 5;
  const int lane = tid & 31;
  const int l16  = lane & 15;
  const int kgrp = lane >> 4;

  const int b     = blockIdx.z;
  const int h     = blockIdx.y;
  const int kvh   = h >> 2;            // head -> kv head (GROUPS = 4)
  const int qbase = blockIdx.x * 64;
  const int qrow0 = qbase + wave * 16;

  // Preload the wave's Q fragments for the whole kernel (4 x 16x32 bf16)
  const u16* qrow = qb + ((long)(b * NH_ + h) * S_ + qrow0 + l16) * HD_;
  u16x16 qf[4];
  #pragma unroll
  for (int kh = 0; kh < 4; kh++) qf[kh] = ld_frag(qrow + kh * 32, kgrp);

  f32x8 O[8];
  #pragma unroll
  for (int f = 0; f < 8; f++)
    #pragma unroll
    for (int i = 0; i < 8; i++) O[f][i] = 0.0f;
  float mrow[8], lrow[8];
  #pragma unroll
  for (int r = 0; r < 8; r++) { mrow[r] = -3.0e38f; lrow[r] = 0.0f; }

  const float scale = 0.08838834764831845f;   // 1/sqrt(128)
  const int ktmax = (qbase + 63) >> 5;        // causal: keys <= last q row

  for (int kt = 0; kt <= ktmax; kt++) {
    const int kbase = kt * 32;
    const u16* ktile = kb + ((long)(b * NKV_ + kvh) * S_ + kbase) * HD_;

#if USE_TDM
    // ---- K tile via Tensor Data Mover: 2-D tile 128(x) x 32(y), bf16 ----
    if (tid == 0) {
      unsigned long long ga = (unsigned long long)(const void*)ktile;
      unsigned ldsa = (unsigned)(unsigned long long)(const void*)&ldsK[0];
      u32x4 g0;
      g0[0] = 1u;                                   // count=1, user descriptor
      g0[1] = ldsa;                                 // lds_addr (bytes)
      g0[2] = (unsigned)ga;                         // global_addr[31:0]
      g0[3] = ((unsigned)(ga >> 32) & 0x01FFFFFFu)  // global_addr[56:32]
              | 0x80000000u;                        // type=2 ("image")
      i32x8 g1;
      g1[0] = 1 << 16;                              // data_size=2B, no mask
      g1[1] = (HD_ & 0xFFFF) << 16;                 // tensor_dim0 = 128
      g1[2] = (S_ & 0xFFFF) << 16;                  // tensor_dim1 = 2048
      g1[3] = HD_ << 16;                            // tile_dim0 = 128
      g1[4] = 32;                                   // tile_dim1 = 32, tile_dim2=0
      g1[5] = HD_;                                  // tensor_dim0_stride = 128
      g1[6] = 0;
      g1[7] = 0;
      i32x4 z4; z4[0] = 0; z4[1] = 0; z4[2] = 0; z4[3] = 0;
      i32x8 z8;
      #pragma unroll
      for (int i = 0; i < 8; i++) z8[i] = 0;
      __builtin_amdgcn_tensor_load_to_lds(g0, g1, z4, z4, z8, 0);
    }
#endif

    // ---- cooperative stage: V^T tile (and K tile on the fallback path) ----
    {
      int key  = tid >> 2;
      int part = tid & 3;
      const u16* vp = vb + ((long)(b * NKV_ + kvh) * S_ + kbase + key) * HD_ + part * 32;
#if !USE_TDM
      const u16* kp = ktile + (long)key * HD_ + part * 32;
#endif
      #pragma unroll
      for (int c = 0; c < 4; c++) {
#if !USE_TDM
        u16x8 kv = *(const u16x8*)(kp + c * 8);
        *(u16x8*)(&ldsK[key * 128 + part * 32 + c * 8]) = kv;
#endif
        u16x8 vv = *(const u16x8*)(vp + c * 8);
        #pragma unroll
        for (int j = 0; j < 8; j++) ldsVt[(part * 32 + c * 8 + j) * 32 + key] = vv[j];
      }
    }
#if USE_TDM
    if (tid == 0) __builtin_amdgcn_s_wait_tensorcnt(0);
#endif
    __syncthreads();

    // ---- scores: S16x32 = Q(16x128) @ K^T(128x32) ----
    f32x8 sacc[2];
    #pragma unroll
    for (int a = 0; a < 2; a++)
      #pragma unroll
      for (int i = 0; i < 8; i++) sacc[a][i] = 0.0f;
    #pragma unroll
    for (int kh = 0; kh < 4; kh++) {
      #pragma unroll
      for (int a = 0; a < 2; a++) {
        u16x16 kf = ld_frag(&ldsK[(a * 16 + l16) * 128 + kh * 32], kgrp);
        sacc[a] = wmma_bf(qf[kh], kf, sacc[a]);
      }
    }

    // ---- online softmax (per-row stats via 16-lane shuffles) ----
    #pragma unroll
    for (int r = 0; r < 8; r++) {
      int m = qrow0 + r + 8 * kgrp;
      #pragma unroll
      for (int a = 0; a < 2; a++) {
        int n = kbase + a * 16 + l16;
        sacc[a][r] = sacc[a][r] * scale + ((n <= m) ? 0.0f : -1.0e9f);
      }
      float rm = fmaxf(sacc[0][r], sacc[1][r]);
      #pragma unroll
      for (int d = 1; d < 16; d <<= 1) rm = fmaxf(rm, __shfl_xor(rm, d, 32));
      float mnew = fmaxf(mrow[r], rm);
      float corr = __expf(mrow[r] - mnew);
      mrow[r] = mnew;
      float psum = 0.0f;
      #pragma unroll
      for (int a = 0; a < 2; a++) {
        float p = __expf(sacc[a][r] - mnew);
        sacc[a][r] = p;
        psum += p;
        ldsP[wave][(r + 8 * kgrp) * 32 + a * 16 + l16] = f2bf(p);
      }
      #pragma unroll
      for (int d = 1; d < 16; d <<= 1) psum += __shfl_xor(psum, d, 32);
      lrow[r] = lrow[r] * corr + psum;
      #pragma unroll
      for (int f = 0; f < 8; f++) O[f][r] *= corr;
    }

    // ---- O += P(16x32) @ V(32x128) ----
    u16x16 pf = ld_frag(&ldsP[wave][l16 * 32], kgrp);
    #pragma unroll
    for (int f = 0; f < 8; f++) {
      u16x16 vf = ld_frag(&ldsVt[(f * 16 + l16) * 32], kgrp);
      O[f] = wmma_bf(pf, vf, O[f]);
    }
    __syncthreads();
  }

  // ---- normalize and write attn output [b][s][h*HD + hd] (f32) ----
  #pragma unroll
  for (int r = 0; r < 8; r++) {
    int m = qrow0 + r + 8 * kgrp;
    float inv = 1.0f / lrow[r];
    #pragma unroll
    for (int f = 0; f < 8; f++)
      attnf[((long)(b * S_) + m) * (NH_ * HD_) + h * HD_ + f * 16 + l16] = O[f][r] * inv;
  }
}

// ---------------------------------------------------------------------------
// 5) attn * sigmoid(gate) -> bf16
// ---------------------------------------------------------------------------
__global__ void gate_pack(const float* __restrict__ attnf, const float* __restrict__ qkv,
                          u16* __restrict__ gb) {
  long idx = (long)blockIdx.x * blockDim.x + threadIdx.x;
  if (idx >= (long)M_ * (NH_ * HD_)) return;
  int r = (int)(idx / (NH_ * HD_)), c = (int)(idx % (NH_ * HD_));
  float g = qkv[(long)r * NQKV_ + 2048 + c];
  float sig = 1.0f / (1.0f + __expf(-g));
  gb[idx] = f2bf(attnf[idx] * sig);
}

// ---------------------------------------------------------------------------
// Launch
// ---------------------------------------------------------------------------
extern "C" void kernel_launch(void* const* d_in, const int* in_sizes, int n_in,
                              void* d_out, int out_size, void* d_ws, size_t ws_size,
                              hipStream_t stream) {
  const float* hidden = (const float*)d_in[0];
  // d_in[1] = attention_mask (pure causal; computed analytically in-kernel)
  const float* Wq  = (const float*)d_in[2];
  const float* Wk  = (const float*)d_in[3];
  const float* Wv  = (const float*)d_in[4];
  const float* Wo  = (const float*)d_in[5];
  const float* qnw = (const float*)d_in[6];
  const float* knw = (const float*)d_in[7];
  float* out = (float*)d_out;

  char* ws = (char*)d_ws;
  size_t off = 0;
  auto carve = [&](size_t bytes) {
    char* p = ws + off;
    off = (off + bytes + 255) & ~(size_t)255;
    return p;
  };

  u16*   hsb    = (u16*)  carve((size_t)M_ * HID_ * 2);           // hidden bf16
  u16*   wqkvb  = (u16*)  carve((size_t)HID_ * NQKV_ * 2);        // [Wq|Wk|Wv] bf16
  u16*   wob    = (u16*)  carve((size_t)HID_ * HID_ * 2);         // Wo bf16
  float* qkv    = (float*)carve((size_t)M_ * NQKV_ * 4);          // projection out
  u16*   qb     = (u16*)  carve((size_t)B_ * NH_  * S_ * HD_ * 2);
  u16*   kb     = (u16*)  carve((size_t)B_ * NKV_ * S_ * HD_ * 2);
  u16*   vb     = (u16*)  carve((size_t)B_ * NKV_ * S_ * HD_ * 2);
  float* attnf  = (float*)carve((size_t)M_ * NH_ * HD_ * 4);
  u16*   gatedb = (u16*)  carve((size_t)M_ * NH_ * HD_ * 2);

  const int TB = 256;
  auto blocks = [](long n, int tb) { return (unsigned)((n + tb - 1) / tb); };

  // 1) converts
  cvt_f32_bf16<<<blocks((long)M_ * HID_, TB), TB, 0, stream>>>(
      hidden, hsb, M_, HID_, HID_, HID_, 0);
  cvt_f32_bf16<<<blocks((long)HID_ * 4096, TB), TB, 0, stream>>>(
      Wq, wqkvb, HID_, 4096, 4096, NQKV_, 0);
  cvt_f32_bf16<<<blocks((long)HID_ * 512, TB), TB, 0, stream>>>(
      Wk, wqkvb, HID_, 512, 512, NQKV_, 4096);
  cvt_f32_bf16<<<blocks((long)HID_ * 512, TB), TB, 0, stream>>>(
      Wv, wqkvb, HID_, 512, 512, NQKV_, 4608);
  cvt_f32_bf16<<<blocks((long)HID_ * HID_, TB), TB, 0, stream>>>(
      Wo, wob, HID_, HID_, HID_, HID_, 0);

  // 2) fused QKV+gate projection: [4096x2048] @ [2048x5120]
  gemm_bf16<<<dim3(NQKV_ / 128, M_ / 64), 256, 0, stream>>>(
      hsb, wqkvb, qkv, M_, NQKV_, HID_, HID_, NQKV_, NQKV_);

  // 3) RMS-norm + RoPE + pack
  normrope_pack<<<dim3(M_, NH_ + 2 * NKV_), 128, 0, stream>>>(
      qkv, qnw, knw, qb, kb, vb);

  // 4) flash attention (TDM-staged K tiles)
  flash_attn<<<dim3(S_ / 64, NH_, B_), 128, 0, stream>>>(qb, kb, vb, attnf);

  // 5) gate + pack
  gate_pack<<<blocks((long)M_ * NH_ * HD_, TB), TB, 0, stream>>>(attnf, qkv, gatedb);

  // 6) output projection: [4096x2048] @ [2048x2048] -> d_out (f32)
  gemm_bf16<<<dim3(HID_ / 128, M_ / 64), 256, 0, stream>>>(
      gatedb, wob, out, M_, HID_, HID_, HID_, HID_, HID_);
}
